// Attention_60816736911748
// MI455X (gfx1250) — matrix operand order repaired and checked by reading, not compiled, not run on hardware
//
#include <hip/hip_runtime.h>
#include <hip/hip_bf16.h>
#include <cstdint>

typedef __attribute__((ext_vector_type(16))) _Float16 v16h;
typedef __attribute__((ext_vector_type(8)))  float    v8f;

#define B_    8
#define N_    4096
#define F_    128
#define U_    64
#define SLOPE 0.2f

// ---------------------------------------------------------------------------
// Phase 1: X = H@W (f32), store X^T as f16 [B][U][N] for WMMA B-operands,
//          s = X@a1, n = X@a2.   (0.5 GFLOP total -- negligible)
// ---------------------------------------------------------------------------
__global__ __launch_bounds__(64)
void gat_phase1(const float* __restrict__ H, const float* __restrict__ W,
                const float* __restrict__ a1, const float* __restrict__ a2,
                _Float16* __restrict__ Xt, float* __restrict__ s,
                float* __restrict__ n) {
  __shared__ float hrow[F_];
  __shared__ float r1[U_], r2[U_];
  const int bj = blockIdx.x;              // b*N + j
  const int b  = bj >> 12;
  const int j  = bj & (N_ - 1);
  const int u  = threadIdx.x;

  const float* Hr = H + (size_t)bj * F_;
  hrow[u]      = Hr[u];
  hrow[u + 64] = Hr[u + 64];
  __syncthreads();

  float acc = 0.f;
  #pragma unroll 8
  for (int f = 0; f < F_; ++f) acc += hrow[f] * W[f * U_ + u];

  Xt[((size_t)(b * U_ + u) << 12) + j] = (_Float16)acc;
  r1[u] = acc * a1[u];
  r2[u] = acc * a2[u];
  __syncthreads();
  for (int w = 32; w > 0; w >>= 1) {
    if (u < w) { r1[u] += r1[u + w]; r2[u] += r2[u + w]; }
    __syncthreads();
  }
  if (u == 0) { s[bj] = r1[0]; n[bj] = r2[0]; }
}

// Phase 1b: nmax[b] = max_j n[b][j]  (for the safe softmax upper bound)
__global__ __launch_bounds__(256)
void gat_nmax(const float* __restrict__ n, float* __restrict__ nmax) {
  __shared__ float red[256];
  const int b = blockIdx.x, t = threadIdx.x;
  float m = -3.0e38f;
  for (int j = t; j < N_; j += 256) m = fmaxf(m, n[b * N_ + j]);
  red[t] = m;
  __syncthreads();
  for (int w = 128; w > 0; w >>= 1) {
    if (t < w) red[t] = fmaxf(red[t], red[t + w]);
    __syncthreads();
  }
  if (t == 0) nmax[b] = red[0];
}

// ---------------------------------------------------------------------------
// Phase 2: fused masked-softmax + (alpha @ X) with v_wmma_f32_16x16x32_f16.
// One wave per 16-row tile; streams A exactly once (512 MB -> ~22 us floor).
// Accumulator tile 4 (all-ones column-0 B tile) carries softmax row sums.
// ---------------------------------------------------------------------------
__global__ __launch_bounds__(32)
void gat_attn(const float* __restrict__ A, const _Float16* __restrict__ Xt,
              const float* __restrict__ s, const float* __restrict__ n,
              const float* __restrict__ nmax, float* __restrict__ out) {
  const int  lane = threadIdx.x;
  const int  b    = blockIdx.y;
  const int  row0 = blockIdx.x * 16;
  const int  m    = lane & 15;
  const bool hi   = lane >= 16;
  const int  row  = row0 + m;

  const float sm  = s[b * N_ + row];
  float bnd = sm + nmax[b];
  bnd = bnd > 0.f ? bnd : bnd * SLOPE;   // lrelu upper bound for this row

  const float*    Arow = A  + ((size_t)(b * N_ + row) << 12);
  const float*    nb   = n  + b * N_;
  const _Float16* Xb   = Xt + ((size_t)(b * U_) << 12);

  v8f acc[4] = {v8f{}, v8f{}, v8f{}, v8f{}};
  v8f accS   = {};

  // B tile with ones in column 0 -> accS column 0 = row sums of P
  union { v16h v; _Float16 h[16]; } onesB;
  {
    _Float16 o = (_Float16)((m == 0) ? 1.0f : 0.0f);
    #pragma unroll
    for (int i = 0; i < 16; ++i) onesB.h[i] = o;
  }

  // A-operand k order used here (contiguous, same for B): lanes 0-15 own K {0..15},
  // lanes 16-31 own K {16..31}, row M = lane&15.
  const int c0 = hi ? 16 : 0;
  const int jb = hi ? 16 : 0;

  for (int j0 = 0; j0 < N_; j0 += 32) {
    const float4* Ac = (const float4*)(Arow + j0 + c0);
    const float4* Nc = (const float4*)(nb   + j0 + c0);
    float4 A0 = Ac[0], A1 = Ac[1], A2 = Ac[2], A3 = Ac[3];   // 16 contiguous floats
    float4 N0 = Nc[0], N1 = Nc[1], N2 = Nc[2], N3 = Nc[3];

    const float av[16] = {A0.x,A0.y,A0.z,A0.w, A1.x,A1.y,A1.z,A1.w,
                          A2.x,A2.y,A2.z,A2.w, A3.x,A3.y,A3.z,A3.w};
    const float nv[16] = {N0.x,N0.y,N0.z,N0.w, N1.x,N1.y,N1.z,N1.w,
                          N2.x,N2.y,N2.z,N2.w, N3.x,N3.y,N3.z,N3.w};

    union { v16h v; _Float16 h[16]; } pa;
    #pragma unroll
    for (int i = 0; i < 16; ++i) {
      float e = sm + nv[i];
      e = e > 0.f ? e : e * SLOPE;                 // leaky-relu
      float p = av[i] * __expf(e - bnd);           // masked entries -> exact 0
      pa.h[i] = (_Float16)p;
    }

    // B operands: Xt rows are contiguous in j; 16 halves = 32 B per lane.
    v16h b0 = *(const v16h*)(Xb + (((size_t)( 0 + m)) << 12) + j0 + jb);
    v16h b1 = *(const v16h*)(Xb + (((size_t)(16 + m)) << 12) + j0 + jb);
    v16h b2 = *(const v16h*)(Xb + (((size_t)(32 + m)) << 12) + j0 + jb);
    v16h b3 = *(const v16h*)(Xb + (((size_t)(48 + m)) << 12) + j0 + jb);

    acc[0] = __builtin_amdgcn_wmma_f32_16x16x32_f16(false, pa.v, false, b0,
                                                    (short)0, acc[0], false, false);
    acc[1] = __builtin_amdgcn_wmma_f32_16x16x32_f16(false, pa.v, false, b1,
                                                    (short)0, acc[1], false, false);
    acc[2] = __builtin_amdgcn_wmma_f32_16x16x32_f16(false, pa.v, false, b2,
                                                    (short)0, acc[2], false, false);
    acc[3] = __builtin_amdgcn_wmma_f32_16x16x32_f16(false, pa.v, false, b3,
                                                    (short)0, acc[3], false, false);
    accS   = __builtin_amdgcn_wmma_f32_16x16x32_f16(false, pa.v, false, onesB.v,
                                                    (short)0, accS, false, false);
  }

  // Normalize + relu + store. C layout: VGPR r -> M=r (lanes 0-15) / M=8+r
  // (lanes 16-31), N = lane&15. Row sum lives in N==0 lane of accS.
  float* outB = out + (size_t)(b * N_ + row0) * U_;
  #pragma unroll
  for (int r = 0; r < 8; ++r) {
    float rs  = __shfl(accS[r], lane & 16, 32);    // bcast lane0 / lane16
    float inv = 1.0f / rs;                         // self-loops => rs > 0
    float* orow = outB + (size_t)(r + (hi ? 8 : 0)) * U_ + m;
    #pragma unroll
    for (int t = 0; t < 4; ++t) {
      float v = acc[t][r] * inv;
      orow[t * 16] = v > 0.f ? v : 0.f;
    }
  }
}

// ---------------------------------------------------------------------------
extern "C" void kernel_launch(void* const* d_in, const int* in_sizes, int n_in,
                              void* d_out, int out_size, void* d_ws, size_t ws_size,
                              hipStream_t stream) {
  const float* H  = (const float*)d_in[0];
  const float* A  = (const float*)d_in[1];
  const float* W  = (const float*)d_in[2];
  const float* a1 = (const float*)d_in[3];
  const float* a2 = (const float*)d_in[4];
  float* out = (float*)d_out;

  // workspace layout: Xt f16 (4 MiB) | s (128 KiB) | n (128 KiB) | nmax (32 B)
  char*     ws   = (char*)d_ws;
  _Float16* Xt   = (_Float16*)ws;
  float*    s    = (float*)(ws + (size_t)B_ * U_ * N_ * sizeof(_Float16));
  float*    n    = s + B_ * N_;
  float*    nmax = n + B_ * N_;

  gat_phase1<<<dim3(B_ * N_), dim3(64), 0, stream>>>(H, W, a1, a2, Xt, s, n);
  gat_nmax<<<dim3(B_), dim3(256), 0, stream>>>(n, nmax);
  gat_attn<<<dim3(N_ / 16, B_), dim3(32), 0, stream>>>(A, Xt, s, n, nmax, out);
}
